// DecoderWithAttention_26809185862161
// MI455X (gfx1250) — matrix operand order repaired and checked by reading, not compiled, not run on hardware
//
#include <hip/hip_runtime.h>
#include <hip/hip_bf16.h>
#include <math.h>

#define BB 128
#define NN 36
#define FF 2048
#define AA 1024
#define EE 1024
#define DD 1024
#define VV 10000
#define LL 20
#define TT 19   // L - 1 decode steps

typedef __bf16 bf16_t;
typedef __attribute__((ext_vector_type(16))) __bf16 v16bf;
typedef __attribute__((ext_vector_type(8)))  __bf16 v8bf;
typedef __attribute__((ext_vector_type(8)))  float  v8f;

// ---------------- WMMA fragment loaders (CDNA5 ISA 7.12.2 layouts) ----------
// A matrix 16x32 bf16 (MxK): lane holds row m=lane&15; lanes 0-15 hold
// K = 0..7 and 16..23, lanes 16-31 hold K = 8..15 and 24..31.
__device__ inline v16bf ldA16(const bf16_t* p) {
  v8bf lo = *(const v8bf*)(p);        // K = khi .. khi+7
  v8bf hi = *(const v8bf*)(p + 16);   // K = khi+16 .. khi+23
  v16bf r;
#pragma unroll
  for (int i = 0; i < 8; ++i) { r[i] = lo[i]; r[i + 8] = hi[i]; }
  return r;
}
// B matrix 32x16 bf16 (KxN): lane holds column n=lane&15; lanes 0-15 hold
// K = 0..7 and 16..23, lanes 16-31 hold K = 8..15 and 24..31 (same as A).
// Since B[k][n] == W[n][k] for C = act @ W^T, these are two 16B runs
// from row n of W.
__device__ inline v16bf ldB16(const bf16_t* p) {
  v8bf lo = *(const v8bf*)(p);
  v8bf hi = *(const v8bf*)(p + 16);
  v16bf r;
#pragma unroll
  for (int i = 0; i < 8; ++i) { r[i] = lo[i]; r[i + 8] = hi[i]; }
  return r;
}

// ---------------- generic bf16 WMMA GEMM: C = A @ W^T (+bias)(+=C)(*mask) ---
// A: (M x K) bf16 row-major, W: (N x K) bf16 row-major, C: f32, row stride ldc.
// flags bit0: accumulate into existing C; bit1: multiply row m by (t < declen[m]).
// NT = compile-time number of 16-wide N tiles per wave.
// K loop is software-pipelined (double-buffered fragments) so next-iteration
// loads are in flight while the current WMMA chain executes.
template <int NT>
__global__ __launch_bounds__(256) void gemm_bf16_wmma(
    const bf16_t* __restrict__ Amat, const bf16_t* __restrict__ Wmat,
    float* __restrict__ C, const float* __restrict__ bias,
    const int* __restrict__ declen,
    int M, int ncol0, int K, int ldc, int t, int flags)
{
  const int wave = (int)(threadIdx.x >> 5);
  const int lane = (int)(threadIdx.x & 31);
  const int mtile = (int)blockIdx.y * 8 + wave;
  if (mtile * 16 >= M) return;                 // wave-uniform: EXEC stays all-1s
  const int nbase = ncol0 + (int)blockIdx.x * (16 * NT);

  const int arow = lane & 15;
  const int akhi = (lane >> 4) * 8;
  const int bkhi = (lane >> 4) * 8;

  const bf16_t* ap = Amat + (size_t)(mtile * 16 + arow) * K + akhi;
  const bf16_t* wp[NT];
#pragma unroll
  for (int j = 0; j < NT; ++j)
    wp[j] = Wmat + (size_t)(nbase + j * 16 + arow) * K + bkhi;

  v8f acc[NT];
#pragma unroll
  for (int j = 0; j < NT; ++j)
#pragma unroll
    for (int r = 0; r < 8; ++r) acc[j][r] = 0.f;

  // ---- software pipeline: prologue loads for k = 0 ----
  v16bf a_cur = ldA16(ap);
  v16bf b_cur[NT];
#pragma unroll
  for (int j = 0; j < NT; ++j) b_cur[j] = ldB16(wp[j]);

#pragma unroll 2
  for (int k = 32; k < K; k += 32) {
    // issue next iteration's loads before consuming current fragments
    v16bf a_nxt = ldA16(ap + k);
    v16bf b_nxt[NT];
#pragma unroll
    for (int j = 0; j < NT; ++j) b_nxt[j] = ldB16(wp[j] + k);
#pragma unroll
    for (int j = 0; j < NT; ++j)
      acc[j] = __builtin_amdgcn_wmma_f32_16x16x32_bf16(
          false, a_cur, false, b_cur[j], (short)0, acc[j], false, false);
    a_cur = a_nxt;
#pragma unroll
    for (int j = 0; j < NT; ++j) b_cur[j] = b_nxt[j];
  }
  // ---- epilogue: last fragment set ----
#pragma unroll
  for (int j = 0; j < NT; ++j)
    acc[j] = __builtin_amdgcn_wmma_f32_16x16x32_bf16(
        false, a_cur, false, b_cur[j], (short)0, acc[j], false, false);

  const int cm = (lane >> 4) * 8;
  const int cn = lane & 15;
#pragma unroll
  for (int j = 0; j < NT; ++j) {
    const int ncol = nbase + j * 16 + cn;
    const float bv = bias ? bias[ncol] : 0.f;
#pragma unroll
    for (int r = 0; r < 8; ++r) {
      const int mrow = mtile * 16 + cm + r;
      float v = acc[j][r] + bv;
      if (flags & 1) v += C[(size_t)mrow * ldc + ncol];
      if (flags & 2) v = (t < declen[mrow]) ? v : 0.f;
      C[(size_t)mrow * ldc + ncol] = v;
    }
  }
}

// ---------------- setup kernels --------------------------------------------
// Stable descending rank sort of 128 lengths (matches jnp.argsort(-lengths)).
__global__ void rank_sort_k(const int* __restrict__ cap_len,
                            int* __restrict__ sort_ind, int* __restrict__ dec_len) {
  __shared__ int sl[BB];
  int i = threadIdx.x;
  sl[i] = cap_len[i];
  __syncthreads();
  int li = sl[i], rank = 0;
  for (int j = 0; j < BB; ++j) {
    int lj = sl[j];
    rank += (lj > li) || (lj == li && j < i);
  }
  sort_ind[rank] = i;
  dec_len[rank]  = li - 1;
}

__global__ void cvt_bf16_k(const float* __restrict__ s, bf16_t* __restrict__ d, int n) {
  int i = blockIdx.x * blockDim.x + threadIdx.x;
  if (i < n) d[i] = (bf16_t)s[i];
}

__global__ void add_bias_k(const float* __restrict__ a, const float* __restrict__ b,
                           float* __restrict__ o, int n) {
  int i = blockIdx.x * blockDim.x + threadIdx.x;
  if (i < n) o[i] = a[i] + b[i];
}

__global__ void gather_feats_k(const float* __restrict__ img, const int* __restrict__ si,
                               bf16_t* __restrict__ feats) {
  int i = blockIdx.x * blockDim.x + threadIdx.x;
  if (i >= BB * NN * FF) return;
  int b = i / (NN * FF);
  int rem = i - b * (NN * FF);
  feats[i] = (bf16_t)img[(size_t)si[b] * (NN * FF) + rem];
}

// mean over N=36 tiles -> write into x1 middle region (cols 1024..3071)
__global__ void feats_mean_k(const bf16_t* __restrict__ feats, bf16_t* __restrict__ x1) {
  int i = blockIdx.x * blockDim.x + threadIdx.x;
  if (i >= BB * FF) return;
  int b = i / FF, f = i - b * FF;
  float s = 0.f;
  for (int n = 0; n < NN; ++n) s += (float)feats[(size_t)(b * NN + n) * FF + f];
  x1[(size_t)b * 4096 + DD + f] = (bf16_t)(s * (1.0f / NN));
}

__global__ void gather_emb_k(const float* __restrict__ emb, const int* __restrict__ caps,
                             const int* __restrict__ si, bf16_t* __restrict__ embs) {
  int i = blockIdx.x * blockDim.x + threadIdx.x;
  if (i >= BB * TT * EE) return;
  int b = i / (TT * EE);
  int rem = i - b * (TT * EE);
  int t = rem / EE, e = rem - t * EE;
  int tok = caps[si[b] * LL + t];
  embs[i] = (bf16_t)emb[(size_t)tok * EE + e];
}

__global__ void zero_states_k(float* h1, float* c1, float* h2, float* c2,
                              bf16_t* h1b, bf16_t* h2b) {
  int i = blockIdx.x * blockDim.x + threadIdx.x;
  if (i >= BB * DD) return;
  h1[i] = 0.f; c1[i] = 0.f; h2[i] = 0.f; c2[i] = 0.f;
  h1b[i] = (bf16_t)0.f; h2b[i] = (bf16_t)0.f;
}

// x1 = [h2 | feats_mean | emb_t]; feats_mean already resident.
__global__ void concat1_k(const float* __restrict__ h2, const bf16_t* __restrict__ embs,
                          bf16_t* __restrict__ x1, int t) {
  int i = blockIdx.x * blockDim.x + threadIdx.x;
  if (i >= BB * DD) return;
  int b = i / DD, j = i - b * DD;
  x1[(size_t)b * 4096 + j]        = (bf16_t)h2[i];
  x1[(size_t)b * 4096 + 3072 + j] = embs[(size_t)(b * TT + t) * EE + j];
}

__device__ inline float sigm(float x) { return 1.f / (1.f + expf(-x)); }

// LSTM cell with active-mask blending; optionally mirrors h into x2 tail region.
__global__ void lstm_cell_k(const float* __restrict__ g, float* __restrict__ h,
                            float* __restrict__ c, bf16_t* __restrict__ hb,
                            bf16_t* __restrict__ x2h, const int* __restrict__ dl, int t) {
  int i = blockIdx.x * blockDim.x + threadIdx.x;
  if (i >= BB * DD) return;
  int b = i / DD, j = i - b * DD;
  const float* gr = g + (size_t)b * 4096;
  float ig = sigm(gr[j]);
  float fg = sigm(gr[DD + j]);
  float gg = tanhf(gr[2 * DD + j]);
  float og = sigm(gr[3 * DD + j]);
  float nc = fg * c[i] + ig * gg;
  float nh = og * tanhf(nc);
  if (t < dl[b]) { c[i] = nc; h[i] = nh; }
  bf16_t hv = (bf16_t)h[i];
  hb[i] = hv;
  if (x2h) x2h[(size_t)b * 3072 + FF + j] = hv;
}

// One block per batch row: e_n = relu(att1+att2).Wa + ba, softmax over 36,
// awe = sum_n alpha_n * feats[b,n,:] -> x2 head region (cols 0..2047).
__global__ __launch_bounds__(256) void attention_k(
    const float* __restrict__ att1, const float* __restrict__ att2,
    const float* __restrict__ Wa, const float* __restrict__ ba,
    const bf16_t* __restrict__ feats, bf16_t* __restrict__ x2)
{
  __shared__ float sm[NN + 4];
  const int b = blockIdx.x;
  const int wave = threadIdx.x >> 5, lane = threadIdx.x & 31;
  for (int n = wave; n < NN; n += 8) {
    const float* a1 = att1 + (size_t)(b * NN + n) * AA;
    const float* a2 = att2 + (size_t)b * AA;
    float p = 0.f;
    for (int a = lane; a < AA; a += 32) {
      float v = a1[a] + a2[a];
      v = v > 0.f ? v : 0.f;
      p += v * Wa[a];
    }
#pragma unroll
    for (int off = 16; off >= 1; off >>= 1) p += __shfl_xor(p, off, 32);
    if (lane == 0) sm[n] = p + ba[0];
  }
  __syncthreads();
  if (threadIdx.x == 0) {
    float mx = sm[0];
    for (int n = 1; n < NN; ++n) mx = sm[n] > mx ? sm[n] : mx;
    float s = 0.f;
    for (int n = 0; n < NN; ++n) { sm[n] = expf(sm[n] - mx); s += sm[n]; }
    float inv = 1.f / s;
    for (int n = 0; n < NN; ++n) sm[n] *= inv;
  }
  __syncthreads();
  for (int f = threadIdx.x; f < FF; f += 256) {
    float s = 0.f;
    for (int n = 0; n < NN; ++n)
      s += (float)feats[(size_t)(b * NN + n) * FF + f] * sm[n];
    x2[(size_t)b * 3072 + f] = (bf16_t)s;
  }
}

// ---------------- host orchestration ---------------------------------------
extern "C" void kernel_launch(void* const* d_in, const int* in_sizes, int n_in,
                              void* d_out, int out_size, void* d_ws, size_t ws_size,
                              hipStream_t stream) {
  (void)in_sizes; (void)n_in; (void)out_size; (void)ws_size;
  const float* image_features = (const float*)d_in[0];
  const int*   caps           = (const int*)d_in[1];
  const int*   cap_len        = (const int*)d_in[2];
  const float* emb            = (const float*)d_in[3];
  const float* W1_ih = (const float*)d_in[4];
  const float* W1_hh = (const float*)d_in[5];
  const float* b1_ih = (const float*)d_in[6];
  const float* b1_hh = (const float*)d_in[7];
  const float* W2_ih = (const float*)d_in[8];
  const float* W2_hh = (const float*)d_in[9];
  const float* b2_ih = (const float*)d_in[10];
  const float* b2_hh = (const float*)d_in[11];
  const float* Wf  = (const float*)d_in[12];
  const float* bf_ = (const float*)d_in[13];
  const float* Wd  = (const float*)d_in[14];
  const float* bd  = (const float*)d_in[15];
  const float* Wa  = (const float*)d_in[16];
  const float* ba  = (const float*)d_in[17];
  const float* Wfc = (const float*)d_in[18];
  const float* bfc = (const float*)d_in[19];
  float* out = (float*)d_out;

  char* base = (char*)d_ws; size_t off = 0;
  auto alloc = [&](size_t bytes) -> void* {
    void* p = base + off;
    off = (off + bytes + 255) & ~(size_t)255;
    return p;
  };
  int*    sort_ind = (int*)alloc(BB * 4);
  int*    dec_len  = (int*)alloc(BB * 4);
  float*  b1c  = (float*)alloc(4096 * 4);
  float*  b2c  = (float*)alloc(4096 * 4);
  bf16_t* feats = (bf16_t*)alloc((size_t)BB * NN * FF * 2);
  float*  att1  = (float*)alloc((size_t)BB * NN * AA * 4);
  bf16_t* embs  = (bf16_t*)alloc((size_t)BB * TT * EE * 2);
  bf16_t* x1    = (bf16_t*)alloc((size_t)BB * 4096 * 2);
  bf16_t* x2    = (bf16_t*)alloc((size_t)BB * 3072 * 2);
  float*  h1 = (float*)alloc((size_t)BB * DD * 4);
  float*  c1 = (float*)alloc((size_t)BB * DD * 4);
  float*  h2 = (float*)alloc((size_t)BB * DD * 4);
  float*  c2 = (float*)alloc((size_t)BB * DD * 4);
  bf16_t* h1b = (bf16_t*)alloc((size_t)BB * DD * 2);
  bf16_t* h2b = (bf16_t*)alloc((size_t)BB * DD * 2);
  float*  g1  = (float*)alloc((size_t)BB * 4096 * 4);
  float*  g2  = (float*)alloc((size_t)BB * 4096 * 4);
  float*  att2 = (float*)alloc((size_t)BB * AA * 4);
  bf16_t* W1ihb = (bf16_t*)alloc((size_t)4096 * 4096 * 2);
  bf16_t* W1hhb = (bf16_t*)alloc((size_t)4096 * 1024 * 2);
  bf16_t* W2ihb = (bf16_t*)alloc((size_t)4096 * 3072 * 2);
  bf16_t* W2hhb = (bf16_t*)alloc((size_t)4096 * 1024 * 2);
  bf16_t* Wfb   = (bf16_t*)alloc((size_t)AA * FF * 2);
  bf16_t* Wdb   = (bf16_t*)alloc((size_t)AA * DD * 2);
  bf16_t* Wfcb  = (bf16_t*)alloc((size_t)VV * DD * 2);

  const int TPB = 256;
  auto blk = [](long n, int t) { return (unsigned)((n + t - 1) / t); };

  // --- setup -----------------------------------------------------------
  hipLaunchKernelGGL(rank_sort_k, dim3(1), dim3(BB), 0, stream, cap_len, sort_ind, dec_len);
  hipLaunchKernelGGL(zero_states_k, dim3(blk(BB * DD, TPB)), dim3(TPB), 0, stream,
                     h1, c1, h2, c2, h1b, h2b);
  hipLaunchKernelGGL(add_bias_k, dim3(blk(4096, TPB)), dim3(TPB), 0, stream, b1_ih, b1_hh, b1c, 4096);
  hipLaunchKernelGGL(add_bias_k, dim3(blk(4096, TPB)), dim3(TPB), 0, stream, b2_ih, b2_hh, b2c, 4096);
  hipLaunchKernelGGL(cvt_bf16_k, dim3(blk((long)4096 * 4096, TPB)), dim3(TPB), 0, stream, W1_ih, W1ihb, 4096 * 4096);
  hipLaunchKernelGGL(cvt_bf16_k, dim3(blk((long)4096 * 1024, TPB)), dim3(TPB), 0, stream, W1_hh, W1hhb, 4096 * 1024);
  hipLaunchKernelGGL(cvt_bf16_k, dim3(blk((long)4096 * 3072, TPB)), dim3(TPB), 0, stream, W2_ih, W2ihb, 4096 * 3072);
  hipLaunchKernelGGL(cvt_bf16_k, dim3(blk((long)4096 * 1024, TPB)), dim3(TPB), 0, stream, W2_hh, W2hhb, 4096 * 1024);
  hipLaunchKernelGGL(cvt_bf16_k, dim3(blk((long)AA * FF, TPB)), dim3(TPB), 0, stream, Wf, Wfb, AA * FF);
  hipLaunchKernelGGL(cvt_bf16_k, dim3(blk((long)AA * DD, TPB)), dim3(TPB), 0, stream, Wd, Wdb, AA * DD);
  hipLaunchKernelGGL(cvt_bf16_k, dim3(blk((long)VV * DD, TPB)), dim3(TPB), 0, stream, Wfc, Wfcb, VV * DD);
  hipLaunchKernelGGL(gather_feats_k, dim3(blk((long)BB * NN * FF, TPB)), dim3(TPB), 0, stream,
                     image_features, sort_ind, feats);
  hipLaunchKernelGGL(feats_mean_k, dim3(blk((long)BB * FF, TPB)), dim3(TPB), 0, stream, feats, x1);
  hipLaunchKernelGGL(gather_emb_k, dim3(blk((long)BB * TT * EE, TPB)), dim3(TPB), 0, stream,
                     emb, caps, sort_ind, embs);

  // att1 = feats @ Wf^T + bf : M=4608, N=1024, K=2048
  hipLaunchKernelGGL((gemm_bf16_wmma<4>), dim3(AA / 64, (BB * NN / 16) / 8), dim3(256), 0, stream,
                     feats, Wfb, att1, bf_, (const int*)nullptr,
                     BB * NN, 0, FF, AA, 0, 0);

  // --- timestep loop ----------------------------------------------------
  for (int t = 0; t < TT; ++t) {
    hipLaunchKernelGGL(concat1_k, dim3(blk(BB * DD, TPB)), dim3(TPB), 0, stream, h2, embs, x1, t);
    // g1 = x1 @ W1_ih^T + (b1_ih + b1_hh)
    hipLaunchKernelGGL((gemm_bf16_wmma<4>), dim3(4096 / 64, 1), dim3(256), 0, stream,
                       x1, W1ihb, g1, b1c, (const int*)nullptr, BB, 0, 4096, 4096, 0, 0);
    // g1 += h1 @ W1_hh^T
    hipLaunchKernelGGL((gemm_bf16_wmma<4>), dim3(4096 / 64, 1), dim3(256), 0, stream,
                       h1b, W1hhb, g1, (const float*)nullptr, (const int*)nullptr,
                       BB, 0, DD, 4096, 0, 1);
    hipLaunchKernelGGL(lstm_cell_k, dim3(blk(BB * DD, TPB)), dim3(TPB), 0, stream,
                       g1, h1, c1, h1b, x2, dec_len, t);
    // att2 = h1 @ Wd^T + bd
    hipLaunchKernelGGL((gemm_bf16_wmma<4>), dim3(AA / 64, 1), dim3(256), 0, stream,
                       h1b, Wdb, att2, bd, (const int*)nullptr, BB, 0, DD, AA, 0, 0);
    hipLaunchKernelGGL(attention_k, dim3(BB), dim3(256), 0, stream,
                       att1, att2, Wa, ba, feats, x2);
    // g2 = x2 @ W2_ih^T + (b2_ih + b2_hh)
    hipLaunchKernelGGL((gemm_bf16_wmma<4>), dim3(4096 / 64, 1), dim3(256), 0, stream,
                       x2, W2ihb, g2, b2c, (const int*)nullptr, BB, 0, 3072, 4096, 0, 0);
    // g2 += h2 @ W2_hh^T
    hipLaunchKernelGGL((gemm_bf16_wmma<4>), dim3(4096 / 64, 1), dim3(256), 0, stream,
                       h2b, W2hhb, g2, (const float*)nullptr, (const int*)nullptr,
                       BB, 0, DD, 4096, 0, 1);
    hipLaunchKernelGGL(lstm_cell_k, dim3(blk(BB * DD, TPB)), dim3(TPB), 0, stream,
                       g2, h2, c2, h2b, (bf16_t*)nullptr, dec_len, t);
    // preds[:, t, :] = (h2 @ Wfc^T + bfc) * active (masked rows -> 0)
    // bulk: 156 blocks x 64 cols = 9984 cols, NT=4
    hipLaunchKernelGGL((gemm_bf16_wmma<4>), dim3(VV / 64, 1), dim3(256), 0, stream,
                       h2b, Wfcb, out + (size_t)t * VV, bfc, dec_len,
                       BB, 0, DD, TT * VV, t, 2);
    // tail: cols 9984..9999, NT=1
    hipLaunchKernelGGL((gemm_bf16_wmma<1>), dim3(1, 1), dim3(256), 0, stream,
                       h2b, Wfcb, out + (size_t)t * VV, bfc, dec_len,
                       BB, (VV / 64) * 64, DD, TT * VV, t, 2);
  }
}
